// MambaBlock_61907658605232
// MI455X (gfx1250) — compile-verified, matrix operand order checked by reading, not run on hardware
//
#include <hip/hip_runtime.h>
#include <hip/hip_bf16.h>
#include <cmath>

#define BATCH  2
#define SEQ    4096
#define DIM    1024
#define NH     16
#define HD     64
#define ROWS   (BATCH*SEQ)      // 8192
#define NCHUNK 64               // SEQ / 64
#define CLEN   64

typedef __bf16 bf16_t;
typedef __attribute__((ext_vector_type(16))) __bf16        v16bf;
typedef __attribute__((ext_vector_type(8)))  float         v8f;
typedef __attribute__((ext_vector_type(4)))  unsigned int  u32x4;

union BFrag { v16bf v; u32x4 q[2]; };

__device__ __forceinline__ float  bf2f(bf16_t x){ return (float)x; }
__device__ __forceinline__ bf16_t f2bf(float x){ return (bf16_t)x; }

// Load one 16x32 bf16 WMMA operand fragment (A layout; B uses the same layout
// when the source matrix is stored transposed, i.e. lane index = output column).
// Lane L (0..15):  row = row0+L, K = k0 + {0..7, 16..23}
// Lane L (16..31): row = row0+L-16, K = k0 + {8..15, 24..31}
__device__ __forceinline__ v16bf load_frag(const bf16_t* __restrict__ base,
                                           int stride, int row0, int k0)
{
  const int lane = threadIdx.x & 31;
  const int r = row0 + (lane & 15);
  const int k = k0 + ((lane >> 4) << 3);
  const bf16_t* p = base + (size_t)r * stride + k;
  BFrag f;
  f.q[0] = *(const u32x4*)(p);
  f.q[1] = *(const u32x4*)(p + 16);
  return f.v;
}

__device__ __forceinline__ v8f wmma_bf16(v16bf a, v16bf b, v8f c)
{
  return __builtin_amdgcn_wmma_f32_16x16x32_bf16(false, a, false, b,
                                                 (short)0, c, false, false);
}

// ---------------------------------------------------------------- prep kernels
__global__ __launch_bounds__(256) void k_prep_w(
    const float* __restrict__ wq, const float* __restrict__ wk,
    const float* __restrict__ wv, const float* __restrict__ wo,
    bf16_t* __restrict__ wqt, bf16_t* __restrict__ wkt,
    bf16_t* __restrict__ wvt, bf16_t* __restrict__ wot)
{
  const float* src; bf16_t* dst;
  switch (blockIdx.y) {
    case 0: src = wq; dst = wqt; break;
    case 1: src = wk; dst = wkt; break;
    case 2: src = wv; dst = wvt; break;
    default: src = wo; dst = wot; break;
  }
  int idx = blockIdx.x * 256 + threadIdx.x;   // over DIM*DIM
  int n = idx >> 10, k = idx & (DIM - 1);
  dst[idx] = f2bf(src[k * DIM + n]);          // Wt[n][k] = W[k][n]
}

__global__ __launch_bounds__(256) void k_prep_x(const float* __restrict__ x,
                                                bf16_t* __restrict__ xb)
{
  int idx = blockIdx.x * 256 + threadIdx.x;   // over ROWS*DIM
  xb[idx] = f2bf(x[idx]);
}

// dt = softplus(x @ w_dt_w + b);  A = -exp(A_log)*dt
__global__ __launch_bounds__(256) void k_dt(
    const float* __restrict__ x, const float* __restrict__ wdt,
    const float* __restrict__ bdt, const float* __restrict__ A_log,
    float* __restrict__ dt, float* __restrict__ Aout)
{
  int idx = blockIdx.x * 256 + threadIdx.x;   // over ROWS*NH
  int row = idx >> 4, h = idx & 15;
  const float* xr = x + (size_t)row * DIM;
  float acc = bdt[h];
  #pragma unroll 8
  for (int d = 0; d < DIM; ++d) acc = fmaf(xr[d], wdt[d * NH + h], acc);
  float dtv = (acc > 20.f) ? acc : log1pf(__expf(acc));
  dt[idx]   = dtv;
  Aout[idx] = -__expf(A_log[h]) * dtv;
}

// ---------------------------------------------------------------- QKV GEMM
// out[m][n] = sum_k xb[m][k] * Wt[n][k].  Each wave computes a 32x64 output
// tile: 2 A-frags + 4 B-frags -> 8 WMMAs per K-step (0.75 loads/WMMA).
__global__ __launch_bounds__(256) void k_gemm_qkv(
    const bf16_t* __restrict__ xb,
    const bf16_t* __restrict__ wqt, const bf16_t* __restrict__ wkt,
    const bf16_t* __restrict__ wvt,
    bf16_t* __restrict__ qb, bf16_t* __restrict__ kb, bf16_t* __restrict__ vb)
{
  const bf16_t* wt  = (blockIdx.z == 0) ? wqt : (blockIdx.z == 1) ? wkt : wvt;
  bf16_t*       out = (blockIdx.z == 0) ? qb  : (blockIdx.z == 1) ? kb  : vb;
  const int wave = threadIdx.x >> 5, lane = threadIdx.x & 31;
  const int m0 = blockIdx.x * 32;
  const int n0 = (blockIdx.y * 8 + wave) * 64;
  v8f acc[2][4] = {};
  for (int k0 = 0; k0 < DIM; k0 += 32) {
    __builtin_prefetch(xb + (size_t)(m0 + (lane & 15)) * DIM + k0 + 256, 0, 3);
    __builtin_prefetch(wt + (size_t)(n0 + (lane & 15)) * DIM + k0 + 256, 0, 3);
    v16bf a0 = load_frag(xb, DIM, m0,      k0);
    v16bf a1 = load_frag(xb, DIM, m0 + 16, k0);
    #pragma unroll
    for (int j = 0; j < 4; ++j) {
      v16bf b = load_frag(wt, DIM, n0 + j * 16, k0);
      acc[0][j] = wmma_bf16(a0, b, acc[0][j]);
      acc[1][j] = wmma_bf16(a1, b, acc[1][j]);
    }
  }
  #pragma unroll
  for (int i = 0; i < 2; ++i) {
    const int mr = m0 + i * 16 + ((lane >> 4) << 3);
    #pragma unroll
    for (int j = 0; j < 4; ++j) {
      const int n = n0 + j * 16 + (lane & 15);
      #pragma unroll
      for (int r = 0; r < 8; ++r)
        out[(size_t)(mr + r) * DIM + n] = f2bf(acc[i][j][r]);
    }
  }
}

// ---------------------------------------------------------------- RoPE + X=V*dt
__global__ __launch_bounds__(256) void k_rope(
    bf16_t* __restrict__ qb, bf16_t* __restrict__ kb, bf16_t* __restrict__ vb,
    const float* __restrict__ cosb, const float* __restrict__ sinb,
    const float* __restrict__ dt)
{
  int idx = blockIdx.x * 256 + threadIdx.x;   // over ROWS*NH*32
  int d   = idx & 31;
  int h   = (idx >> 5) & 15;
  int row = idx >> 9;
  int l   = row & (SEQ - 1);
  size_t base = (size_t)row * DIM + h * HD;
  float c1 = cosb[l * HD + d],      s1 = sinb[l * HD + d];
  float c2 = cosb[l * HD + d + 32], s2 = sinb[l * HD + d + 32];
  float q1 = bf2f(qb[base + d]), q2 = bf2f(qb[base + d + 32]);
  qb[base + d]      = f2bf(q1 * c1 - q2 * s1);
  qb[base + d + 32] = f2bf(q2 * c2 + q1 * s2);
  float k1 = bf2f(kb[base + d]), k2 = bf2f(kb[base + d + 32]);
  kb[base + d]      = f2bf(k1 * c1 - k2 * s1);
  kb[base + d + 32] = f2bf(k2 * c2 + k1 * s2);
  float dv = dt[row * NH + h];
  vb[base + d]      = f2bf(bf2f(vb[base + d]) * dv);
  vb[base + d + 32] = f2bf(bf2f(vb[base + d + 32]) * dv);
}

// ---------------------------------------------------------------- SSD stage 1
// Per (b,h,chunk): cumsum(A), G=C*B^T (WMMA), decay-mask, Y_diag=G'*X (WMMA),
// chunk states = (decayed B)^T * X (WMMA).
__global__ __launch_bounds__(256) void k_ssd1(
    const bf16_t* __restrict__ qb, const bf16_t* __restrict__ kb,
    const bf16_t* __restrict__ vb, const float* __restrict__ A,
    float* __restrict__ acs_g, float* __restrict__ states,
    float* __restrict__ yf)
{
  __shared__ float  s_acs[CLEN];
  __shared__ bf16_t s_g  [CLEN * CLEN];   // masked/decayed G, A-operand layout
  __shared__ bf16_t s_xT [CLEN * CLEN];   // X transposed: [p][t]
  __shared__ bf16_t s_bdT[CLEN * CLEN];   // (B*decay) transposed: [n][t]

  const int bid = blockIdx.x;                   // (b*NH + h)*NCHUNK + c
  const int c = bid & 63, h = (bid >> 6) & 15, b = bid >> 10;
  const int l0 = b * SEQ + c * CLEN;            // row base into (ROWS,DIM)
  const int col0 = h * HD;
  const int tid = threadIdx.x, lane = tid & 31, wave = tid >> 5;

  if (tid == 0) {
    float s = 0.f;
    for (int t = 0; t < CLEN; ++t) { s += A[(size_t)(l0 + t) * NH + h]; s_acs[t] = s; }
  }
  __syncthreads();
  const float atot = s_acs[CLEN - 1];
  if (tid < CLEN) acs_g[(size_t)bid * CLEN + tid] = s_acs[tid];

  for (int idx = tid; idx < CLEN * CLEN; idx += 256) {
    int t = idx >> 6, p = idx & 63;
    s_xT[p * CLEN + t]  = f2bf(bf2f(vb[(size_t)(l0 + t) * DIM + col0 + p]));
    float bvv = bf2f(kb[(size_t)(l0 + t) * DIM + col0 + p]);
    s_bdT[p * CLEN + t] = f2bf(bvv * __expf(atot - s_acs[t]));
  }

  // G = C @ B^T over K = state-dim; 16 tiles, 2 per wave.
  int ti[2], tj[2];
  v8f g[2];
  #pragma unroll
  for (int s = 0; s < 2; ++s) {
    int id = wave * 2 + s; ti[s] = id >> 2; tj[s] = id & 3;
    v8f acc = {};
    for (int kk = 0; kk < CLEN; kk += 32) {
      v16bf a  = load_frag(qb, DIM, l0 + ti[s] * 16, col0 + kk);
      v16bf bb = load_frag(kb, DIM, l0 + tj[s] * 16, col0 + kk);
      acc = wmma_bf16(a, bb, acc);
    }
    g[s] = acc;
  }
  // mask + decay, restage into LDS as next A-operand
  #pragma unroll
  for (int s = 0; s < 2; ++s) {
    int jb  = tj[s] * 16 + (lane & 15);
    int ib0 = ti[s] * 16 + ((lane >> 4) << 3);
    #pragma unroll
    for (int r = 0; r < 8; ++r) {
      int i = ib0 + r;
      float v = (i >= jb) ? g[s][r] * __expf(s_acs[i] - s_acs[jb]) : 0.f;
      s_g[i * CLEN + jb] = f2bf(v);
    }
  }
  __syncthreads();

  // Y_diag = G' @ X   and   states = Bd^T @ X   (both K = 64, from LDS)
  #pragma unroll
  for (int s = 0; s < 2; ++s) {
    v8f acc = {};
    for (int kk = 0; kk < CLEN; kk += 32) {
      v16bf a  = load_frag(s_g,  CLEN, ti[s] * 16, kk);
      v16bf bb = load_frag(s_xT, CLEN, tj[s] * 16, kk);
      acc = wmma_bf16(a, bb, acc);
    }
    int pcol  = col0 + tj[s] * 16 + (lane & 15);
    int irow0 = l0 + ti[s] * 16 + ((lane >> 4) << 3);
    #pragma unroll
    for (int r = 0; r < 8; ++r) yf[(size_t)(irow0 + r) * DIM + pcol] = acc[r];
  }
  #pragma unroll
  for (int s = 0; s < 2; ++s) {
    v8f acc = {};
    for (int kk = 0; kk < CLEN; kk += 32) {
      v16bf a  = load_frag(s_bdT, CLEN, ti[s] * 16, kk);
      v16bf bb = load_frag(s_xT,  CLEN, tj[s] * 16, kk);
      acc = wmma_bf16(a, bb, acc);
    }
    int n0 = ti[s] * 16 + ((lane >> 4) << 3);
    int p  = tj[s] * 16 + (lane & 15);
    float* stp = states + (size_t)bid * (CLEN * CLEN);
    #pragma unroll
    for (int r = 0; r < 8; ++r) stp[(n0 + r) * CLEN + p] = acc[r];
  }
}

// ---------------------------------------------------------------- SSD stage 2
// Sequential inter-chunk scan per (b,h); rewrites states[] in place with the
// *incoming* state per chunk and writes the final state to d_out tail.
__global__ __launch_bounds__(256) void k_ssd2(
    float* __restrict__ states, const float* __restrict__ acs_g,
    float* __restrict__ state_out)
{
  const int bh = blockIdx.x;      // 0..31
  const int tid = threadIdx.x;
  float run[16];
  #pragma unroll
  for (int k = 0; k < 16; ++k) run[k] = 0.f;
  for (int c = 0; c < NCHUNK; ++c) {
    float* st = states + ((size_t)bh * NCHUNK + c) * (CLEN * CLEN);
    float dec = __expf(acs_g[((size_t)bh * NCHUNK + c) * CLEN + (CLEN - 1)]);
    #pragma unroll
    for (int k = 0; k < 16; ++k) {
      int e = tid + k * 256;
      float v = st[e];
      st[e]  = run[k];            // incoming state for chunk c
      run[k] = run[k] * dec + v;
    }
  }
  #pragma unroll
  for (int k = 0; k < 16; ++k) {
    int e = tid + k * 256;
    int n = e >> 6, p = e & 63;   // buffer is (n,p); output is (b,h,p,n)
    state_out[((size_t)bh * 64 + p) * 64 + n] = run[k];
  }
}

// ---------------------------------------------------------------- SSD stage 3
// Y_off = diag(exp(acs)) * (C @ state_in); Y = Y_diag + Y_off -> bf16 y.
__global__ __launch_bounds__(256) void k_ssd3(
    const bf16_t* __restrict__ qb, const float* __restrict__ states,
    const float* __restrict__ acs_g, const float* __restrict__ yf,
    bf16_t* __restrict__ yb)
{
  __shared__ bf16_t s_stT[CLEN * CLEN];   // state transposed: [p][n]
  const int bid = blockIdx.x;
  const int c = bid & 63, h = (bid >> 6) & 15, b = bid >> 10;
  const int l0 = b * SEQ + c * CLEN, col0 = h * HD;
  const int tid = threadIdx.x, lane = tid & 31, wave = tid >> 5;

  const float* st = states + (size_t)bid * (CLEN * CLEN);
  for (int idx = tid; idx < CLEN * CLEN; idx += 256) {
    int n = idx >> 6, p = idx & 63;
    s_stT[p * CLEN + n] = f2bf(st[idx]);
  }
  __syncthreads();

  const float* acs = acs_g + (size_t)bid * CLEN;
  #pragma unroll
  for (int s = 0; s < 2; ++s) {
    int id = wave * 2 + s; int ti = id >> 2, tp = id & 3;
    v8f acc = {};
    for (int kk = 0; kk < CLEN; kk += 32) {
      v16bf a  = load_frag(qb, DIM, l0 + ti * 16, col0 + kk);
      v16bf bb = load_frag(s_stT, CLEN, tp * 16, kk);
      acc = wmma_bf16(a, bb, acc);
    }
    int pcol = col0 + tp * 16 + (lane & 15);
    int i0   = ti * 16 + ((lane >> 4) << 3);
    #pragma unroll
    for (int r = 0; r < 8; ++r) {
      int i = i0 + r;
      float y = yf[(size_t)(l0 + i) * DIM + pcol] + __expf(acs[i]) * acc[r];
      yb[(size_t)(l0 + i) * DIM + pcol] = f2bf(y);
    }
  }
}

// ---------------------------------------------------------------- output GEMM
__global__ __launch_bounds__(256) void k_gemm_out(
    const bf16_t* __restrict__ yb, const bf16_t* __restrict__ wot,
    float* __restrict__ out)
{
  const int wave = threadIdx.x >> 5, lane = threadIdx.x & 31;
  const int m0 = blockIdx.x * 32;
  const int n0 = (blockIdx.y * 8 + wave) * 64;
  v8f acc[2][4] = {};
  for (int k0 = 0; k0 < DIM; k0 += 32) {
    __builtin_prefetch(yb  + (size_t)(m0 + (lane & 15)) * DIM + k0 + 256, 0, 3);
    __builtin_prefetch(wot + (size_t)(n0 + (lane & 15)) * DIM + k0 + 256, 0, 3);
    v16bf a0 = load_frag(yb, DIM, m0,      k0);
    v16bf a1 = load_frag(yb, DIM, m0 + 16, k0);
    #pragma unroll
    for (int j = 0; j < 4; ++j) {
      v16bf b = load_frag(wot, DIM, n0 + j * 16, k0);
      acc[0][j] = wmma_bf16(a0, b, acc[0][j]);
      acc[1][j] = wmma_bf16(a1, b, acc[1][j]);
    }
  }
  #pragma unroll
  for (int i = 0; i < 2; ++i) {
    const int mr = m0 + i * 16 + ((lane >> 4) << 3);
    #pragma unroll
    for (int j = 0; j < 4; ++j) {
      const int n = n0 + j * 16 + (lane & 15);
      #pragma unroll
      for (int r = 0; r < 8; ++r)
        out[(size_t)(mr + r) * DIM + n] = acc[i][j][r];
    }
  }
}

// ---------------------------------------------------------------- launch
extern "C" void kernel_launch(void* const* d_in, const int* in_sizes, int n_in,
                              void* d_out, int out_size, void* d_ws, size_t ws_size,
                              hipStream_t stream) {
  const float* x     = (const float*)d_in[0];
  const float* cosb  = (const float*)d_in[1];
  const float* sinb  = (const float*)d_in[2];
  const float* wq    = (const float*)d_in[3];
  const float* wk    = (const float*)d_in[4];
  const float* wv    = (const float*)d_in[5];
  const float* wo    = (const float*)d_in[6];
  const float* A_log = (const float*)d_in[7];
  const float* wdtw  = (const float*)d_in[8];
  const float* wdtb  = (const float*)d_in[9];

  char* ws = (char*)d_ws;
  bf16_t* xb  = (bf16_t*)ws;                 ws += (size_t)ROWS * DIM * 2;
  bf16_t* wqt = (bf16_t*)ws;                 ws += (size_t)DIM * DIM * 2;
  bf16_t* wkt = (bf16_t*)ws;                 ws += (size_t)DIM * DIM * 2;
  bf16_t* wvt = (bf16_t*)ws;                 ws += (size_t)DIM * DIM * 2;
  bf16_t* wot = (bf16_t*)ws;                 ws += (size_t)DIM * DIM * 2;
  bf16_t* qb  = (bf16_t*)ws;                 ws += (size_t)ROWS * DIM * 2;
  bf16_t* kb  = (bf16_t*)ws;                 ws += (size_t)ROWS * DIM * 2;
  bf16_t* vb  = (bf16_t*)ws;                 ws += (size_t)ROWS * DIM * 2;
  float*  dt  = (float*)ws;                  ws += (size_t)ROWS * NH * 4;
  float*  Aa  = (float*)ws;                  ws += (size_t)ROWS * NH * 4;
  float*  acs = (float*)ws;                  ws += (size_t)BATCH * NH * NCHUNK * CLEN * 4;
  float*  st  = (float*)ws;                  ws += (size_t)BATCH * NH * NCHUNK * CLEN * CLEN * 4;
  float*  yf  = (float*)ws;                  ws += (size_t)ROWS * DIM * 4;
  bf16_t* yb  = (bf16_t*)ws;                 ws += (size_t)ROWS * DIM * 2;

  float* out       = (float*)d_out;
  float* state_out = out + (size_t)ROWS * DIM;

  k_prep_w <<<dim3(4096, 4), 256, 0, stream>>>(wq, wk, wv, wo, wqt, wkt, wvt, wot);
  k_prep_x <<<dim3((ROWS * DIM) / 256), 256, 0, stream>>>(x, xb);
  k_dt     <<<dim3((ROWS * NH) / 256), 256, 0, stream>>>(x, wdtw, wdtb, A_log, dt, Aa);
  k_gemm_qkv<<<dim3(ROWS / 32, DIM / 512, 3), 256, 0, stream>>>(xb, wqt, wkt, wvt, qb, kb, vb);
  k_rope   <<<dim3((ROWS * NH * 32) / 256), 256, 0, stream>>>(qb, kb, vb, cosb, sinb, dt);
  k_ssd1   <<<dim3(BATCH * NH * NCHUNK), 256, 0, stream>>>(qb, kb, vb, Aa, acs, st, yf);
  k_ssd2   <<<dim3(BATCH * NH), 256, 0, stream>>>(st, acs, state_out);
  k_ssd3   <<<dim3(BATCH * NH * NCHUNK), 256, 0, stream>>>(qb, st, acs, yf, yb);
  k_gemm_out<<<dim3(ROWS / 32, DIM / 512), 256, 0, stream>>>(yb, wot, out);
}
